// GeneralizedWindowAttention_15281493639411
// MI455X (gfx1250) — compile-verified, matrix operand order checked by reading, not run on hardware
//
#include <hip/hip_runtime.h>

// ---------------------------------------------------------------------------
// Generalized window attention, fully fused per-window on gfx1250 (wave32).
// One workgroup (128 threads = 4 waves) per window b. All GEMMs via
// v_wmma_f32_16x16x32_f16; f32 accumulation; softmax/bias in f32 VALU.
// Round 3: hoist r-invariant epilogue operands (qkb/pb/Bw) out of writeout
// loops and pre-gather bott[] rows, so LDS loads batch ahead of the b16
// stores instead of interleaved load/store/wait chains.
// ---------------------------------------------------------------------------

#define DIMC 96
#define NH   3
#define EDIM 32
#define WSZ  7
#define KT   49          // tokens per window
#define MP   64          // padded tokens (4 x 16)
#define NTHREADS 128
#define NWAVES   4
#define QKSCALE  0.17677669529663687f   // 32^-0.5

typedef __attribute__((ext_vector_type(16))) _Float16 v16h;
typedef __attribute__((ext_vector_type(8)))  float    v8f;

// LDS layout. Stage-2 weight staging overlaps stage-3 attention scratch.
struct Smem {
  _Float16 x_h[MP * DIMC];                 // x[b], f16, rows >= KT zeroed
  union {
    struct { _Float16 qkw[192 * DIMC]; float qkb[192]; } s2;     // 37632 B
    struct { float S[MP * MP];                                   // attn logits
             _Float16 attn[MP * MP];                             // softmaxed
             _Float16 qxh[MP * DIMC]; } s3;                      // (attn@x)*bott
  } u;
  _Float16 q_h[MP * DIMC];                 // scaled q, all heads
  _Float16 k_h[MP * DIMC];                 // k, all heads
  _Float16 val_h[MP * DIMC];               // values (+vbias), all heads
  _Float16 Gw[DIMC * DIMC];                // G_w as [c][h*32+e]
  _Float16 pw[DIMC * DIMC];                // proj_w [out][in]
  float pb[DIMC];
  float Bw[DIMC];
  float bott[MP];
};

// K index of element j (0..15) of an A/B fragment for lane-group g (0/1).
// ISA 7.12.2, 16-bit A 16x32: VGPR0..3 -> K {0..7}+8g, VGPR4..7 -> K {16..23}+8g.
__device__ __forceinline__ int frag_k(int j, int g) {
  int v = j >> 1, p = j & 1;
  int base = (v < 4) ? (2 * v) : (16 + 2 * (v - 4));
  return base + 8 * g + p;
}

// A fragment: 16x32 tile at (row0, col0) of row-major buf (leading dim ld).
__device__ __forceinline__ v16h load_a(const _Float16* buf, int ld, int row0, int col0) {
  int lane = threadIdx.x & 31;
  int m = lane & 15, g = lane >> 4;
  v16h a;
#pragma unroll
  for (int j = 0; j < 16; ++j)
    a[j] = buf[(row0 + m) * ld + col0 + frag_k(j, g)];
  return a;
}

// B fragment, source stored row-major as [K][N]: B(k,n) = buf[(k0+k)*ld + n0+n].
__device__ __forceinline__ v16h load_b_rm(const _Float16* buf, int ld, int k0, int n0) {
  int lane = threadIdx.x & 31;
  int n = lane & 15, g = lane >> 4;
  v16h b;
#pragma unroll
  for (int j = 0; j < 16; ++j)
    b[j] = buf[(k0 + frag_k(j, g)) * ld + n0 + n];
  return b;
}

// B fragment, source stored [N][K] (e.g. weights [out][in]): B(k,n) = buf[(n0+n)*ld + k0+k].
__device__ __forceinline__ v16h load_b_cm(const _Float16* buf, int ld, int k0, int n0) {
  int lane = threadIdx.x & 31;
  int n = lane & 15, g = lane >> 4;
  v16h b;
#pragma unroll
  for (int j = 0; j < 16; ++j)
    b[j] = buf[(n0 + n) * ld + k0 + frag_k(j, g)];
  return b;
}

__device__ __forceinline__ v8f wmma(v16h a, v16h b, v8f c) {
  return __builtin_amdgcn_wmma_f32_16x16x32_f16(false, a, false, b, (short)0, c,
                                                false, false);
}

__global__ void gwa_fused_kernel(
    const float* __restrict__ x, const float* __restrict__ mask,
    const float* __restrict__ qk_w, const float* __restrict__ qk_b,
    const float* __restrict__ rpb, const float* __restrict__ bott_w,
    const float* __restrict__ bott_b, const float* __restrict__ G_w,
    const float* __restrict__ B_w, const float* __restrict__ proj_w,
    const float* __restrict__ proj_b, const int* __restrict__ rp_index,
    float* __restrict__ out, int nW) {
  extern __shared__ char smem_raw[];
  Smem& sm = *reinterpret_cast<Smem*>(smem_raw);

  const int b    = blockIdx.x;
  const int tid  = threadIdx.x;
  const int wave = tid >> 5;
  const int lane = tid & 31;
  const int cn   = lane & 15;       // C-fragment column within tile
  const int cg   = lane >> 4;       // C-fragment row group

  const float* xb    = x + (size_t)b * KT * DIMC;
  const float* maskw = mask + (size_t)(b % nW) * KT * KT;

  // ---- Stage 0: global -> LDS (f32 -> f16) ------------------------------
  for (int i = tid; i < MP * DIMC; i += NTHREADS) {
    int r = i / DIMC, c = i - r * DIMC;
    sm.x_h[i] = (_Float16)(r < KT ? xb[r * DIMC + c] : 0.0f);
  }
  for (int i = tid; i < 192 * DIMC; i += NTHREADS) sm.u.s2.qkw[i] = (_Float16)qk_w[i];
  for (int i = tid; i < 192; i += NTHREADS)        sm.u.s2.qkb[i] = qk_b[i];
  for (int i = tid; i < DIMC * DIMC; i += NTHREADS) {
    sm.Gw[i] = (_Float16)G_w[i];           // G_w flat == [c][h*32+e]
    sm.pw[i] = (_Float16)proj_w[i];
  }
  for (int i = tid; i < DIMC; i += NTHREADS) { sm.pb[i] = proj_b[i]; sm.Bw[i] = B_w[i]; }

  // ---- Stage 1: bottleneck scalar per token (f32 dot) -------------------
  for (int t = tid; t < MP; t += NTHREADS) {
    float acc = 0.0f;
    if (t < KT) {
      acc = bott_b[0];
      for (int c = 0; c < DIMC; ++c) acc += xb[t * DIMC + c] * bott_w[c];
    }
    sm.bott[t] = acc;
  }
  __syncthreads();

  // ---- Stage 2: QK projection  [64 x 192] = x_h @ qk_w^T ----------------
  // N-tiles 0..5 are entirely q-columns (n < 96), 6..11 entirely k-columns,
  // so the q/k split is wave-uniform per tile (no exec divergence).
  for (int t = wave; t < 4 * 12; t += NWAVES) {
    int mt = t / 12, nt = t - mt * 12;
    v8f acc = {};
#pragma unroll
    for (int ks = 0; ks < DIMC; ks += 32) {
      v16h a  = load_a(sm.x_h, DIMC, mt * 16, ks);
      v16h bf = load_b_cm(sm.u.s2.qkw, DIMC, ks, nt * 16);
      acc = wmma(a, bf, acc);
    }
    int n = nt * 16 + cn;
    const float biasn = sm.u.s2.qkb[n];    // hoisted: r-invariant
    if (nt < 6) {                          // q tile (uniform branch)
#pragma unroll
      for (int r = 0; r < 8; ++r) {
        int m = mt * 16 + r + 8 * cg;
        sm.q_h[m * DIMC + n] = (_Float16)((acc[r] + biasn) * QKSCALE);
      }
    } else {                               // k tile (uniform branch)
      int nk = n - DIMC;
#pragma unroll
      for (int r = 0; r < 8; ++r) {
        int m = mt * 16 + r + 8 * cg;
        sm.k_h[m * DIMC + nk] = (_Float16)(acc[r] + biasn);
      }
    }
  }
  __syncthreads();

  // ---- Stage 3: per-head attention ---------------------------------------
  for (int h = 0; h < NH; ++h) {
    // 3a: logits S[64][64] = q_h(:,h) @ k_h(:,h)^T  (+bias +mask, pad=-inf)
    for (int t = wave; t < 16; t += NWAVES) {
      int mt = t >> 2, nt = t & 3;
      v16h a  = load_a(sm.q_h, DIMC, mt * 16, h * EDIM);
      v16h bf = load_b_cm(sm.k_h, DIMC, h * EDIM, nt * 16);
      v8f acc = {};
      acc = wmma(a, bf, acc);
      int n = nt * 16 + cn;
      if (mt < 3 && nt < 3) {
        // interior tile: all (m,n) < 49, uniform fast path
#pragma unroll
        for (int r = 0; r < 8; ++r) {
          int m = mt * 16 + r + 8 * cg;
          int ridx = rp_index[m * KT + n];
          sm.u.s3.S[m * MP + n] =
              acc[r] + rpb[ridx * NH + h] + maskw[m * KT + n];
        }
      } else {
        // edge tile: guard per element
#pragma unroll
        for (int r = 0; r < 8; ++r) {
          int m = mt * 16 + r + 8 * cg;
          float v;
          if (m < KT && n < KT) {
            int ridx = rp_index[m * KT + n];
            v = acc[r] + rpb[ridx * NH + h] + maskw[m * KT + n];
          } else {
            v = -1.0e30f;
          }
          sm.u.s3.S[m * MP + n] = v;
        }
      }
    }
    __syncthreads();

    // 3b: row softmax (f32) -> attn f16, padded cols/rows zero
    if (tid < MP) {
      if (tid < KT) {
        float* Sr = &sm.u.s3.S[tid * MP];
        float mx = -1.0e30f;
        for (int j = 0; j < KT; ++j) mx = fmaxf(mx, Sr[j]);
        float sum = 0.0f;
        for (int j = 0; j < KT; ++j) { float e = __expf(Sr[j] - mx); Sr[j] = e; sum += e; }
        float inv = 1.0f / sum;
        for (int j = 0; j < KT; ++j) sm.u.s3.attn[tid * MP + j] = (_Float16)(Sr[j] * inv);
        for (int j = KT; j < MP; ++j) sm.u.s3.attn[tid * MP + j] = (_Float16)0.0f;
      } else {
        for (int j = 0; j < MP; ++j) sm.u.s3.attn[tid * MP + j] = (_Float16)0.0f;
      }
    }
    __syncthreads();

    // 3c: qx[64][96] = attn @ x_h, row-scaled by bott -> qxh (f16)
    for (int t = wave; t < 4 * 6; t += NWAVES) {
      int mt = t / 6, nt = t - mt * 6;
      v8f acc = {};
#pragma unroll
      for (int ks = 0; ks < MP; ks += 32) {
        v16h a  = load_a(sm.u.s3.attn, MP, mt * 16, ks);
        v16h bf = load_b_rm(sm.x_h, DIMC, ks, nt * 16);
        acc = wmma(a, bf, acc);
      }
      int n = nt * 16 + cn;
      float bm[8];                          // pre-gather: loads batch before stores
#pragma unroll
      for (int r = 0; r < 8; ++r) bm[r] = sm.bott[mt * 16 + r + 8 * cg];
#pragma unroll
      for (int r = 0; r < 8; ++r) {
        int m = mt * 16 + r + 8 * cg;
        sm.u.s3.qxh[m * DIMC + n] = (_Float16)(acc[r] * bm[r]);
      }
    }
    __syncthreads();

    // 3d: values[64][32] = qxh @ Gr[:,h,:]  (+ bott*B_w vbias) -> val_h cols
    for (int t = wave; t < 4 * 2; t += NWAVES) {
      int mt = t >> 1, nt = t & 1;
      v8f acc = {};
#pragma unroll
      for (int ks = 0; ks < DIMC; ks += 32) {
        v16h a  = load_a(sm.u.s3.qxh, DIMC, mt * 16, ks);
        v16h bf = load_b_rm(sm.Gw, DIMC, ks, h * EDIM + nt * 16);
        acc = wmma(a, bf, acc);
      }
      int ncol = h * EDIM + nt * 16 + cn;
      const float bwn = sm.Bw[ncol];        // hoisted: r-invariant
      float bm[8];
#pragma unroll
      for (int r = 0; r < 8; ++r) bm[r] = sm.bott[mt * 16 + r + 8 * cg];
#pragma unroll
      for (int r = 0; r < 8; ++r) {
        int m = mt * 16 + r + 8 * cg;
        sm.val_h[m * DIMC + ncol] = (_Float16)(acc[r] + bm[r] * bwn);
      }
    }
    __syncthreads();
  }

  // ---- Stage 4: output projection  out = val @ proj_w^T + proj_b --------
  for (int t = wave; t < 4 * 6; t += NWAVES) {
    int mt = t / 6, nt = t - mt * 6;
    v8f acc = {};
#pragma unroll
    for (int ks = 0; ks < DIMC; ks += 32) {
      v16h a  = load_a(sm.val_h, DIMC, mt * 16, ks);
      v16h bf = load_b_cm(sm.pw, DIMC, ks, nt * 16);
      acc = wmma(a, bf, acc);
    }
    int n = nt * 16 + cn;
    const float pbn = sm.pb[n];             // hoisted: r-invariant
    float* ob = out + (size_t)b * KT * DIMC;
    if (mt < 3) {
      // full tile: all rows < 49, uniform store path
#pragma unroll
      for (int r = 0; r < 8; ++r) {
        int m = mt * 16 + r + 8 * cg;
        ob[m * DIMC + n] = acc[r] + pbn;
      }
    } else {
#pragma unroll
      for (int r = 0; r < 8; ++r) {
        int m = mt * 16 + r + 8 * cg;
        if (m < KT) ob[m * DIMC + n] = acc[r] + pbn;
      }
    }
  }
}

extern "C" void kernel_launch(void* const* d_in, const int* in_sizes, int n_in,
                              void* d_out, int out_size, void* d_ws, size_t ws_size,
                              hipStream_t stream) {
  const float* x      = (const float*)d_in[0];
  const float* mask   = (const float*)d_in[1];
  const float* qk_w   = (const float*)d_in[2];
  const float* qk_b   = (const float*)d_in[3];
  const float* rpb    = (const float*)d_in[4];
  const float* bott_w = (const float*)d_in[5];
  const float* bott_b = (const float*)d_in[6];
  const float* G_w    = (const float*)d_in[7];
  const float* B_w    = (const float*)d_in[8];
  const float* proj_w = (const float*)d_in[9];
  const float* proj_b = (const float*)d_in[10];
  const int*   rp_idx = (const int*)d_in[11];
  float* out = (float*)d_out;

  int BW = in_sizes[0] / (KT * DIMC);      // 128 windows
  int nW = in_sizes[1] / (KT * KT);        // 64 masks

  size_t shmem = sizeof(Smem);             // ~122 KB, fits 320 KB/WGP LDS
  gwa_fused_kernel<<<dim3(BW), dim3(NTHREADS), shmem, stream>>>(
      x, mask, qk_w, qk_b, rpb, bott_w, bott_b, G_w, B_w, proj_w, proj_b,
      rp_idx, out, nW);
}
